// MultiHeadAttention_90933047591397
// MI455X (gfx1250) — matrix operand order repaired and checked by reading, not compiled, not run on hardware
//
#include <hip/hip_runtime.h>
#include <hip/hip_bf16.h>
#include <stdint.h>

// ---------------- problem constants (from reference) ----------------
#define BB 4
#define SS 4096
#define DD 1024
#define HH 16
#define LL 256          // S / HEAD_N
#define KC 32           // K-chunk per WMMA step (bf16 16x16x32)
#define BM 128          // block tile M
#define BN 128          // block tile N  (8 waves: wave tile 32x64)

typedef __bf16 bf16_t;
typedef __bf16 bf16x2  __attribute__((ext_vector_type(2)));
typedef __bf16 bf16x4  __attribute__((ext_vector_type(4)));
typedef __bf16 bf16x8  __attribute__((ext_vector_type(8)));
typedef __bf16 bf16x16 __attribute__((ext_vector_type(16)));
typedef float  f32x4   __attribute__((ext_vector_type(4)));
typedef float  f32x8   __attribute__((ext_vector_type(8)));

// ------------- gfx1250 async global->LDS path (inline asm) ----------------
// ISA 15.18 VGLOBAL: GLOBAL_LOAD_ASYNC_TO_LDS_B128, GV addressing:
//   global_load_async_to_lds_b128 vdst_lds, v[addr:addr+1], off
// vdst holds the 32-bit LDS byte address (low 32 bits of the flat pointer,
// per the ISA flat->LDS truncation rule). Tracked by ASYNCcnt.
__device__ __forceinline__ void cp_b128(const bf16_t* g, bf16_t* l) {
  uint32_t lds_addr = (uint32_t)(uintptr_t)l;
  uint64_t gaddr    = (uint64_t)(uintptr_t)g;
  asm volatile("global_load_async_to_lds_b128 %0, %1, off"
               :
               : "v"(lds_addr), "v"(gaddr)
               : "memory");
}
#define ASYNC_WAIT() asm volatile("s_wait_asynccnt 0x0" ::: "memory")

__device__ __forceinline__ bf16x16 cat16(bf16x8 lo, bf16x8 hi) {
  return __builtin_shufflevector(lo, hi, 0, 1, 2, 3, 4, 5, 6, 7,
                                 8, 9, 10, 11, 12, 13, 14, 15);
}

// One K-step of the 8-wave 128x128 tile: each wave owns a 32x64 patch as
// 2x4 v_wmma_f32_16x16x32_bf16. As = row-major [BM][KC], Bs = transposed
// [BN][KC] (Bs[n][k]). Fragment addressing follows the CDNA5 16-bit A/B VGPR
// layouts (A: lanes 0-15 K{0..7,16..23}, lanes 16-31 K{8..15,24..31};
// B: lanes 0-15 K{0..7,16..23}, lanes 16-31 K{8..15,24..31}, N = lane&15).
__device__ __forceinline__ void mma_step8(const bf16_t* As, const bf16_t* Bs,
                                          int wm, int wn, int lane,
                                          f32x8 acc[2][4]) {
  const int row = lane & 15;
  const int kh  = lane >> 4;
  bf16x16 a[2], b[4];
#pragma unroll
  for (int t = 0; t < 2; ++t) {
    const bf16_t* pa = As + (size_t)(wm + t * 16 + row) * KC;
    a[t] = cat16(*(const bf16x8*)(pa + kh * 8),
                 *(const bf16x8*)(pa + 16 + kh * 8));
  }
#pragma unroll
  for (int t = 0; t < 4; ++t) {
    const bf16_t* pb = Bs + (size_t)(wn + t * 16 + row) * KC + kh * 8;
    b[t] = cat16(*(const bf16x8*)(pb), *(const bf16x8*)(pb + 16));
  }
#pragma unroll
  for (int i = 0; i < 2; ++i)
#pragma unroll
    for (int j = 0; j < 4; ++j)
      acc[i][j] = __builtin_amdgcn_wmma_f32_16x16x32_bf16(
          false, a[i], false, b[j], (short)0, acc[i][j], false, false);
}

template <typename OT>
__device__ __forceinline__ void store_acc(f32x8 acc[2][4], OT* out,
                                          size_t o_stride, int m0, int n0,
                                          int wm, int wn, int lane, float scale) {
  const int orow = (lane >> 4) * 8;  // C/D layout: lanes 16-31 hold M=8..15
  const int ocol = lane & 15;
#pragma unroll
  for (int i = 0; i < 2; ++i)
#pragma unroll
    for (int j = 0; j < 4; ++j)
#pragma unroll
      for (int r = 0; r < 8; ++r)
        out[(size_t)(m0 + wm + i * 16 + orow + r) * o_stride +
            (n0 + wn + j * 16 + ocol)] = (OT)(acc[i][j][r] * scale);
}

// Core tile loop: both A and Bt are bf16 with contiguous 64B K-slices per row,
// so staging is pure async b128 copies (4 lanes per row).
template <typename OT>
__device__ __forceinline__ void gemm_tile(const bf16_t* __restrict__ A, size_t a_stride,
                                          const bf16_t* __restrict__ Bt, size_t b_stride,
                                          int K, OT* __restrict__ out, size_t o_stride,
                                          float scale, int m0, int n0,
                                          bf16_t* As, bf16_t* Bs) {
  const int tid = threadIdx.x, lane = tid & 31, wave = tid >> 5;
  const int wm = (wave & 3) * 32, wn = (wave >> 2) * 64;
  const int r  = tid >> 2;           // 0..63
  const int c8 = (tid & 3) * 8;      // bf16 offset in row: 0,8,16,24

  f32x8 acc[2][4] = {};
  for (int k0 = 0; k0 < K; k0 += KC) {
#pragma unroll
    for (int p = 0; p < 2; ++p) {
      const int row = r + p * 64;
      cp_b128(&A [(size_t)(m0 + row) * a_stride + k0 + c8], &As[row * KC + c8]);
      cp_b128(&Bt[(size_t)(n0 + row) * b_stride + k0 + c8], &Bs[row * KC + c8]);
    }
    ASYNC_WAIT();
    __syncthreads();
    mma_step8(As, Bs, wm, wn, lane, acc);
    __syncthreads();
  }
  store_acc(acc, out, o_stride, m0, n0, wm, wn, lane, scale);
}

// ---------------- generic bf16 GEMM: out = A @ Bt^T ------------------------
template <typename OT>
__global__ __launch_bounds__(256) void gemm_bf16(const bf16_t* __restrict__ A, size_t a_stride,
                                                 const bf16_t* __restrict__ Bt, size_t b_stride,
                                                 int K, OT* __restrict__ out, size_t o_stride,
                                                 float scale) {
  __shared__ __align__(16) bf16_t As[BM * KC];
  __shared__ __align__(16) bf16_t Bs[BN * KC];
  gemm_tile<OT>(A, a_stride, Bt, b_stride, K, out, o_stride, scale,
                blockIdx.y * BM, blockIdx.x * BN, As, Bs);
}

// ------------- score = (Qh @ Kh^T) / sqrt(D), per (b,h) -------------------
// grid (LL/BN, LL/BM, B*H); head rows are strided by H*D in the flat layout.
__global__ __launch_bounds__(256) void score_kernel(const bf16_t* __restrict__ Q,
                                                    const bf16_t* __restrict__ Km,
                                                    float* __restrict__ score) {
  __shared__ __align__(16) bf16_t As[BM * KC];
  __shared__ __align__(16) bf16_t Bs[BN * KC];
  const int bh = blockIdx.z, b = bh >> 4, h = bh & 15;
  const size_t rs = (size_t)HH * DD;
  gemm_tile<float>(Q  + ((size_t)b * SS + h) * DD, rs,
                   Km + ((size_t)b * SS + h) * DD, rs,
                   DD, score + (size_t)bh * LL * LL, LL,
                   0.03125f /* 1/sqrt(1024) */,
                   blockIdx.y * BM, blockIdx.x * BN, As, Bs);
}

// ------------- mask-fill + softmax, one block per score row --------------
__global__ __launch_bounds__(256) void softmax_mask_kernel(
    const float* __restrict__ score, const int* __restrict__ mask,
    bf16_t* __restrict__ att) {
  const int rid = blockIdx.x;       // (b*H + h)*L + l
  const int m = threadIdx.x;        // 256 threads, one per column
  const int l = rid & (LL - 1);
  const int b = rid >> 12;          // rid / (H*L)

  float v = score[(size_t)rid * LL + m];
  if (mask[((size_t)b * LL + l) * LL + m] == 0) v = 1e-9f;  // MASK_FILL

  __shared__ float redmax[8], redsum[8];
  float x = v;
#pragma unroll
  for (int o = 16; o > 0; o >>= 1) x = fmaxf(x, __shfl_xor(x, o, 32));
  if ((m & 31) == 0) redmax[m >> 5] = x;
  __syncthreads();
  float mx = redmax[0];
#pragma unroll
  for (int i = 1; i < 8; ++i) mx = fmaxf(mx, redmax[i]);

  float e = __expf(v - mx);
  float s = e;
#pragma unroll
  for (int o = 16; o > 0; o >>= 1) s += __shfl_xor(s, o, 32);
  if ((m & 31) == 0) redsum[m >> 5] = s;
  __syncthreads();
  float tot = 0.f;
#pragma unroll
  for (int i = 0; i < 8; ++i) tot += redsum[i];

  att[(size_t)rid * LL + m] = (bf16_t)(e / tot);
}

// ------------- x_h = att @ Vh, scattered back to (B,S,D) ------------------
// grid (DD/BN, LL/BM, B*H). A rows are contiguous (async); B needs transpose.
__global__ __launch_bounds__(256) void attv_kernel(const bf16_t* __restrict__ att,
                                                   const bf16_t* __restrict__ V,
                                                   bf16_t* __restrict__ x) {
  __shared__ __align__(16) bf16_t As[BM * KC];
  __shared__ __align__(16) bf16_t Bs[BN * KC];
  const int tid = threadIdx.x, lane = tid & 31, wave = tid >> 5;
  const int wm = (wave & 3) * 32, wn = (wave >> 2) * 64;
  const int m0 = blockIdx.y * BM;   // l
  const int n0 = blockIdx.x * BN;   // d
  const int bh = blockIdx.z, b = bh >> 4, h = bh & 15;
  const size_t rs = (size_t)HH * DD;
  const bf16_t* Ab = att + (size_t)bh * LL * LL;
  const bf16_t* Vb = V + ((size_t)b * SS + h) * DD;
  bf16_t* xb = x + ((size_t)b * SS + h) * DD;
  const int r = tid >> 2, c8 = (tid & 3) * 8;

  f32x8 acc[2][4] = {};
  for (int k0 = 0; k0 < LL; k0 += KC) {
#pragma unroll
    for (int p = 0; p < 2; ++p) {
      const int row = r + p * 64;
      cp_b128(&Ab[(size_t)(m0 + row) * LL + k0 + c8], &As[row * KC + c8]);
    }
    // transpose-stage Vh[k][n] -> Bs[n][k]; coalesced bf16x2 loads along d
#pragma unroll
    for (int idx = tid; idx < (KC * BN) / 2; idx += 256) {
      int kk = idx >> 6, nn = (idx & 63) * 2;
      bf16x2 vv = *(const bf16x2*)&Vb[(size_t)(k0 + kk) * rs + (n0 + nn)];
      Bs[(nn + 0) * KC + kk] = vv.x;
      Bs[(nn + 1) * KC + kk] = vv.y;
    }
    ASYNC_WAIT();
    __syncthreads();
    mma_step8(As, Bs, wm, wn, lane, acc);
    __syncthreads();
  }
  // row l scatters to global row b*S + l*H + h  (stride H*D)
  store_acc(acc, xb, rs, m0, n0, wm, wn, lane, 1.0f);
}

// ---------------- one-shot fp32 -> bf16 conversion (bandwidth pass) -------
__global__ __launch_bounds__(256) void cvt_kernel(const f32x4* __restrict__ in,
                                                  bf16x4* __restrict__ out, int n4) {
  int i = blockIdx.x * 256 + threadIdx.x;
  if (i < n4) out[i] = __builtin_convertvector(in[i], bf16x4);
}

// -------- weight transpose+convert: Wt[n][k] = bf16(W[k][n]), 1024x1024 ---
__global__ __launch_bounds__(256) void wt_kernel(const float* __restrict__ W,
                                                 bf16_t* __restrict__ Wt) {
  __shared__ float tile[32][33];
  const int bx = blockIdx.x * 32;   // n
  const int by = blockIdx.y * 32;   // k
  const int tx = threadIdx.x & 31, ty = threadIdx.x >> 5;
#pragma unroll
  for (int yy = ty; yy < 32; yy += 8)
    tile[yy][tx] = W[(size_t)(by + yy) * DD + bx + tx];
  __syncthreads();
#pragma unroll
  for (int yy = ty; yy < 32; yy += 8)
    Wt[(size_t)(bx + yy) * DD + by + tx] = (bf16_t)tile[tx][yy];
}

// --------------------------- launcher -------------------------------------
extern "C" void kernel_launch(void* const* d_in, const int* in_sizes, int n_in,
                              void* d_out, int out_size, void* d_ws,
                              size_t ws_size, hipStream_t stream) {
  const float* q    = (const float*)d_in[0];
  const float* k    = (const float*)d_in[1];
  const float* v    = (const float*)d_in[2];
  const int*   mask = (const int*)d_in[3];
  const float* Wq   = (const float*)d_in[4];
  const float* Wk   = (const float*)d_in[5];
  const float* Wv   = (const float*)d_in[6];
  const float* Wo   = (const float*)d_in[7];
  float* out = (float*)d_out;

  const size_t nQ = (size_t)BB * SS * DD;        // 16,777,216
  const size_t nW = (size_t)DD * DD;             //  1,048,576

  // workspace layout (regions recycled after the projections consume them)
  bf16_t* qbf = (bf16_t*)d_ws;      // bf16 inputs
  bf16_t* kbf = qbf + nQ;
  bf16_t* vbf = kbf + nQ;
  bf16_t* Qbf = vbf + nQ;           // projected heads
  bf16_t* Kbf = Qbf + nQ;
  bf16_t* Vbf = Kbf + nQ;
  bf16_t* Wqt = Vbf + nQ;           // transposed bf16 weights
  bf16_t* Wkt = Wqt + nW;
  bf16_t* Wvt = Wkt + nW;
  bf16_t* Wot = Wvt + nW;
  // recycled regions (each fully rewritten before use, same order every call):
  float*  score = (float*)qbf;      // 16 MB in 32 MB region
  bf16_t* att   = kbf;              //  8 MB in 32 MB region
  bf16_t* xbf   = vbf;              // 32 MB region

  const dim3 blk(256);
  const dim3 gCvt((unsigned)(nQ / 4 / 256));
  const dim3 gWt(DD / 32, DD / 32);
  const dim3 gProj(DD / BN, (BB * SS) / BM);      // (8, 128)
  const dim3 gScore(LL / BN, LL / BM, BB * HH);   // (2, 2, 64)
  const dim3 gAttV(DD / BN, LL / BM, BB * HH);    // (8, 2, 64)

  // 0) one-shot precision conversion (bandwidth-bound)
  cvt_kernel<<<gCvt, blk, 0, stream>>>((const f32x4*)q, (bf16x4*)qbf, (int)(nQ / 4));
  cvt_kernel<<<gCvt, blk, 0, stream>>>((const f32x4*)k, (bf16x4*)kbf, (int)(nQ / 4));
  cvt_kernel<<<gCvt, blk, 0, stream>>>((const f32x4*)v, (bf16x4*)vbf, (int)(nQ / 4));
  wt_kernel<<<gWt, blk, 0, stream>>>(Wq, Wqt);
  wt_kernel<<<gWt, blk, 0, stream>>>(Wk, Wkt);
  wt_kernel<<<gWt, blk, 0, stream>>>(Wv, Wvt);
  wt_kernel<<<gWt, blk, 0, stream>>>(Wo, Wot);

  // 1) projections (pure bf16 WMMA, fp32 accumulate)
  gemm_bf16<bf16_t><<<gProj, blk, 0, stream>>>(qbf, DD, Wqt, DD, DD, Qbf, DD, 1.0f);
  gemm_bf16<bf16_t><<<gProj, blk, 0, stream>>>(kbf, DD, Wkt, DD, DD, Kbf, DD, 1.0f);
  gemm_bf16<bf16_t><<<gProj, blk, 0, stream>>>(vbf, DD, Wvt, DD, DD, Vbf, DD, 1.0f);

  // 2) attention: scores, mask+softmax, att @ V
  score_kernel<<<gScore, blk, 0, stream>>>(Qbf, Kbf, score);
  softmax_mask_kernel<<<dim3(BB * HH * LL), blk, 0, stream>>>(score, mask, att);
  attv_kernel<<<gAttV, blk, 0, stream>>>(att, Vbf, xbf);

  // 3) output projection (bf16 in, fp32 out)
  gemm_bf16<float><<<gProj, blk, 0, stream>>>(xbf, DD, Wot, DD, DD, out, DD, 1.0f);
}
